// OuterProductMean_53867479826840
// MI455X (gfx1250) — matrix operand order repaired and checked by reading, not compiled, not run on hardware
//
#include <hip/hip_runtime.h>
#include <hip/hip_bf16.h>

// ---------------------------------------------------------------------------
// OuterProductMean fused WMMA implementation for gfx1250 (wave32, WMMA).
//   S=64 MSA rows, T=384 tokens, CM=64, C=32 hidden, CZ=128 out channels.
// Stage A (prep): LayerNorm + masked projections -> a,b in f16 scratch.
// Stage B (opm):  per 16x16 token tile, fused GEMM3 (outer) + GEMM4 (w_out).
// Stage-4 wave blocking: 4 M-tiles x 4 N-tiles per wave so each w_out
// B-fragment (global) and each outer A-fragment (LDS) feeds 4 WMMAs.
// ---------------------------------------------------------------------------

typedef __attribute__((ext_vector_type(16))) _Float16 v16h;
typedef __attribute__((ext_vector_type(8)))  _Float16 v8h;
typedef __attribute__((ext_vector_type(8)))  float    v8f;

#define NS   64     // N_msa
#define NT   384    // N_tok
#define CMD  64     // c_m
#define CHD  32     // c_h
#define CZD  128    // c_z
#define CCH  4      // c-values per chunk (K-chunk of stage 4 = 128)
#define KC   (CCH*CHD)   // 128

// ---------------- prep: LayerNorm + projections (one wave per (s,i) row) ----
__global__ __launch_bounds__(256) void opm_prep_kernel(
    const float* __restrict__ m, const float* __restrict__ mask,
    const float* __restrict__ lnw, const float* __restrict__ lnb,
    const float* __restrict__ w1, const float* __restrict__ w2,
    _Float16* __restrict__ a_scr,   // [c=32][i=384][s=64]
    _Float16* __restrict__ b_scr)   // [j=384][d=32][s=64]
{
  const int wave = (blockIdx.x * blockDim.x + threadIdx.x) >> 5;
  const int lane = threadIdx.x & 31;
  if (wave >= NS * NT) return;
  const int s = wave / NT;
  const int i = wave - s * NT;

  const float* row = m + ((size_t)s * NT + i) * CMD;
  float x0 = row[lane];
  float x1 = row[lane + 32];
  float sum = x0 + x1, sq = x0 * x0 + x1 * x1;
  #pragma unroll
  for (int o = 16; o > 0; o >>= 1) {
    sum += __shfl_xor(sum, o);
    sq  += __shfl_xor(sq,  o);
  }
  const float mu   = sum * (1.0f / 64.0f);
  const float var  = sq * (1.0f / 64.0f) - mu * mu;
  const float rstd = rsqrtf(var + 1e-5f);
  const float n0 = (x0 - mu) * rstd * lnw[lane]      + lnb[lane];
  const float n1 = (x1 - mu) * rstd * lnw[lane + 32] + lnb[lane + 32];
  const float msk = mask[(size_t)s * NT + i];

  // lane computes hidden channel c = lane for both projections
  const float* w1r = w1 + lane * CMD;
  const float* w2r = w2 + lane * CMD;
  float acca = 0.f, accb = 0.f;
  #pragma unroll
  for (int mm = 0; mm < 32; ++mm) {
    float v0 = __shfl(n0, mm);
    float v1 = __shfl(n1, mm);
    acca += v0 * w1r[mm] + v1 * w1r[mm + 32];
    accb += v0 * w2r[mm] + v1 * w2r[mm + 32];
  }
  acca *= msk;
  accb *= msk;
  a_scr[((size_t)lane * NT + i) * NS + s]  = (_Float16)acca;
  b_scr[((size_t)i * CHD + lane) * NS + s] = (_Float16)accb;
}

// ---------------- w_out f32 -> f16 ------------------------------------------
__global__ __launch_bounds__(256) void opm_cvt_wout(
    const float* __restrict__ w, _Float16* __restrict__ o, int n)
{
  int t = blockIdx.x * blockDim.x + threadIdx.x;
  if (t < n) o[t] = (_Float16)w[t];
}

// ---------------- fused outer-product + w_out contraction -------------------
__global__ __launch_bounds__(256) void opm_main_kernel(
    const _Float16* __restrict__ a_scr,  // [c][i][s]
    const _Float16* __restrict__ b_scr,  // [j][d][s]
    const _Float16* __restrict__ wo,     // [z][k=c*32+d] f16
    const float* __restrict__ mask,
    const float* __restrict__ b_out,
    float* __restrict__ out)             // [i][j][z]
{
  // LDS: B tile (512 jd-rows x 64 s) 64KB, outer chunk (256 pairs x 128 k) 64KB
  __shared__ __align__(16) _Float16 Blds[512 * NS];
  __shared__ __align__(16) _Float16 Olds[256 * KC];
  __shared__ float Ninv[256];

  const int i0 = blockIdx.x * 16;
  const int j0 = blockIdx.y * 16;
  const int tid  = threadIdx.x;
  const int wv   = tid >> 5;
  const int lane = tid & 31;
  const int hl   = lane >> 4;   // lane half (0/1)
  const int l16  = lane & 15;

  // stage-4 blocking: wave -> 4 consecutive M-tiles x 4 consecutive N-tiles
  const int mtg = wv >> 1;      // 0..3  -> mt = mtg*4 + mi
  const int ntg = wv & 1;       // 0..1  -> nt = ntg*4 + ni

  // ---- cooperative copy of b tile: contiguous 32768 f16 from b_scr
  {
    const uint4* src = (const uint4*)(b_scr + (size_t)j0 * CHD * NS);
    uint4* dst = (uint4*)Blds;
    #pragma unroll 1
    for (int t = tid; t < (512 * NS) / 8; t += 256) dst[t] = src[t];
  }
  // ---- mask norm reciprocals for the 256 pairs
  {
    const int ii = i0 + (tid >> 4);
    const int jj = j0 + (tid & 15);
    float nrm = 0.f;
    #pragma unroll 1
    for (int s = 0; s < NS; ++s)
      nrm += mask[s * NT + ii] * mask[s * NT + jj];
    Ninv[tid] = 1.0f / (nrm + 1e-3f);
  }

  v8f acc[4][4];
  {
    v8f z = {};
    #pragma unroll
    for (int u = 0; u < 4; ++u)
      #pragma unroll
      for (int v = 0; v < 4; ++v) acc[u][v] = z;
  }

  __syncthreads();

  #pragma unroll 1
  for (int ch = 0; ch < CHD / CCH; ++ch) {
    const int c0 = ch * CCH;

    // ================= stage 3: outer chunk into Olds ======================
    // wave wv handles c_local = wv&3, N-tile half (wv>>2)
    const int cl = wv & 3;
    const int nthalf = wv >> 2;
    {
      // A fragment: rows (c, i), K = s.  Per ISA A-layout: lane needs two
      // contiguous 8-f16 K runs at hl*8 and 16+hl*8 within each 32-K step.
      const _Float16* arow =
          a_scr + ((size_t)(c0 + cl) * NT + i0 + l16) * NS;
      union { v16h v; v8h h[2]; } Af[2];
      #pragma unroll
      for (int ks = 0; ks < 2; ++ks) {
        Af[ks].h[0] = *(const v8h*)(arow + ks * 32 + hl * 8);
        Af[ks].h[1] = *(const v8h*)(arow + ks * 32 + 16 + hl * 8);
      }
      #pragma unroll 1
      for (int t = 0; t < 16; ++t) {
        const int nt = nthalf * 16 + t;
        // B fragment: column n = nt*16+l16, two 8-f16 K runs at hl*8 and 16+hl*8
        const _Float16* brow = Blds + (size_t)(nt * 16 + l16) * NS;
        union { v16h v; v8h h[2]; } Bf[2];
        #pragma unroll
        for (int ks = 0; ks < 2; ++ks) {
          Bf[ks].h[0] = *(const v8h*)(brow + ks * 32 + hl * 8);
          Bf[ks].h[1] = *(const v8h*)(brow + ks * 32 + 16 + hl * 8);
        }
        v8f d = {};
        d = __builtin_amdgcn_wmma_f32_16x16x32_f16(
            false, Af[0].v, false, Bf[0].v, (short)0, d, false, false);
        d = __builtin_amdgcn_wmma_f32_16x16x32_f16(
            false, Af[1].v, false, Bf[1].v, (short)0, d, false, false);
        // scatter D tile into Olds[p = i*16+j][k = cl*32 + d] as f16
        const int n  = nt * 16 + l16;
        const int jl = n >> 5;
        const int dd = n & 31;
        #pragma unroll
        for (int v = 0; v < 8; ++v) {
          const int iy = v + 8 * hl;   // D row m
          Olds[(size_t)(iy * 16 + jl) * KC + cl * CHD + dd] = (_Float16)d[v];
        }
      }
    }
    __syncthreads();

    // ================= stage 4: pairs x z, K = 128 chunk ===================
    #pragma unroll 1
    for (int ks = 0; ks < KC / 32; ++ks) {
      // A fragments for this wave's 4 M-tiles (from LDS outer chunk)
      union { v16h v; v8h h[2]; } A4[4];
      #pragma unroll
      for (int mi = 0; mi < 4; ++mi) {
        const int mt = mtg * 4 + mi;   // i_local
        const _Float16* orow =
            Olds + (size_t)(mt * 16 + l16) * KC + ks * 32;
        A4[mi].h[0] = *(const v8h*)(orow + hl * 8);
        A4[mi].h[1] = *(const v8h*)(orow + 16 + hl * 8);
      }
      #pragma unroll 1
      for (int ni = 0; ni < 4; ++ni) {
        const int nt = ntg * 4 + ni;
        // B fragment straight from global (L2-resident w_out f16);
        // reused by 4 WMMAs below.
        const _Float16* wrow = wo + (size_t)(nt * 16 + l16) * (CHD * CHD)
                               + ch * KC + ks * 32 + hl * 8;
        union { v16h v; v8h h[2]; } B4;
        B4.h[0] = *(const v8h*)(wrow);
        B4.h[1] = *(const v8h*)(wrow + 16);
        #pragma unroll
        for (int mi = 0; mi < 4; ++mi) {
          acc[mi][ni] = __builtin_amdgcn_wmma_f32_16x16x32_f16(
              false, A4[mi].v, false, B4.v, (short)0, acc[mi][ni], false, false);
        }
      }
    }
    __syncthreads();
  }

  // ---------------- epilogue: + b_out, * 1/(norm+eps), store ----------------
  #pragma unroll 1
  for (int mi = 0; mi < 4; ++mi) {
    const int mt = mtg * 4 + mi;         // i_local
    const int ii = i0 + mt;
    #pragma unroll 1
    for (int ni = 0; ni < 4; ++ni) {
      const int nt = ntg * 4 + ni;
      const int z  = nt * 16 + l16;
      const float bz = b_out[z];
      #pragma unroll
      for (int v = 0; v < 8; ++v) {
        const int jl = v + 8 * hl;       // j_local
        const float r = (acc[mi][ni][v] + bz) * Ninv[mt * 16 + jl];
        out[((size_t)ii * NT + j0 + jl) * CZD + z] = r;
      }
    }
  }
}

// ---------------------------------------------------------------------------
extern "C" void kernel_launch(void* const* d_in, const int* in_sizes, int n_in,
                              void* d_out, int out_size, void* d_ws, size_t ws_size,
                              hipStream_t stream) {
  const float* m     = (const float*)d_in[0];
  const float* mask  = (const float*)d_in[1];
  const float* lnw   = (const float*)d_in[2];
  const float* lnb   = (const float*)d_in[3];
  const float* w1    = (const float*)d_in[4];
  const float* w2    = (const float*)d_in[5];
  const float* w_out = (const float*)d_in[6];
  const float* b_out = (const float*)d_in[7];
  float* out = (float*)d_out;

  char* ws = (char*)d_ws;
  const size_t ab_bytes = (size_t)CHD * NT * NS * sizeof(_Float16); // 1.5 MB
  _Float16* a_scr = (_Float16*)ws;
  _Float16* b_scr = (_Float16*)(ws + ab_bytes);
  _Float16* wo    = (_Float16*)(ws + 2 * ab_bytes);                // 256 KB

  // LayerNorm + projections: one wave per (s,i) row -> 24576 waves
  opm_prep_kernel<<<(NS * NT) / 8, 256, 0, stream>>>(
      m, mask, lnw, lnb, w1, w2, a_scr, b_scr);

  // w_out -> f16
  opm_cvt_wout<<<(CZD * CHD * CHD) / 256, 256, 0, stream>>>(
      w_out, wo, CZD * CHD * CHD);

  // fused outer-product-mean + output projection
  dim3 grid(NT / 16, NT / 16);
  opm_main_kernel<<<grid, 256, 0, stream>>>(a_scr, b_scr, wo, mask, b_out, out);
}
